// MultiLabelContrastiveFocalLoss_58918361366660
// MI455X (gfx1250) — matrix operand order repaired and checked by reading, not compiled, not run on hardware
//
#include <hip/hip_runtime.h>
#include <hip/hip_bf16.h>
#include <cstddef>
#include <cstdint>

// ---------------------------------------------------------------------------
// MultiLabelContrastiveFocalLoss for MI455X (gfx1250)
//
//   out = mean(ALPHA*(1-p)^2 * bce)
//       + [ (||s||^2 - Sp) - (||T^T P||_F^2 - Spt) ] / (B*(B-1))
//
//   p   = sigmoid(x);  s_k = sum_i p_ik ;  Sp = sum_i ||p_i||^2
//   Spt = sum_i ||t_i||^2 * ||p_i||^2      (diagonal correction, t in {0,1})
//
//   ||T^T P||_F^2 via bf16 WMMA (16x16x32) with exact-T + p = hi + lo split.
//   Pipeline:
//     convert: one pass over X,T -> bf16 tiles in WMMA fragment-major order
//              (also accumulates column sums)
//     gemm:    LDS double-buffered, staged with global_load_async_to_lds_b128
//              (ASYNCcnt), fragments read as contiguous ds_load_b128 pairs.
//
//   Workspace requirement: ~50.4 MB (3 x 16.78 MB bf16 panels + 16 KB stats).
// ---------------------------------------------------------------------------

#define ALPHA_C 0.25f
static constexpr int Bn = 4096;   // batch
static constexpr int Ln = 2048;   // labels
static constexpr int KC = 32;     // K-chunk (i-dimension)
static constexpr int NCHUNK = Bn / KC;          // 128
static constexpr int NTILE  = Ln / 16;          // 128 tiles per chunk
static constexpr int TILE_E = 512;              // 16x32 bf16 elems per tile

typedef __attribute__((ext_vector_type(16))) __bf16 v16bf;
typedef __attribute__((ext_vector_type(8)))  float  v8f;

// workspace layout (bytes):
//   [0,64)         : 8 doubles  — [0] focal sum, [1] Sp, [2] Spt, [4] frob
//   [64,64+8192)   : 2048 f32 column sums s_k
//   [16384, +16MB) : TbA  bf16 [chunk][tile][512]  (A-swizzled targets)
//   next 16MB      : Phi  bf16 (B-swizzled sigmoid hi)
//   next 16MB      : Plo  bf16 (B-swizzled sigmoid lo)
static constexpr int    WS_FLOATS = 16 + Ln;
static constexpr size_t PANEL_E   = (size_t)NCHUNK * NTILE * TILE_E;  // bf16 elems
static constexpr size_t OFF_TB    = 16384;
static constexpr size_t OFF_PH    = OFF_TB + PANEL_E * 2;
static constexpr size_t OFF_PL    = OFF_PH + PANEL_E * 2;

__global__ __launch_bounds__(256) void zero_ws_kernel(float* __restrict__ ws) {
    int i = blockIdx.x * 256 + threadIdx.x;
    if (i < WS_FLOATS) ws[i] = 0.0f;
}

// ---------------------------------------------------------------------------
// One pass over X,T: emit fragment-major bf16 panels + column sums of p.
// ISA per-lane layouts (16-bit operands, 16x16x32):
//   A (16Mx32K): lane = m + 16*((k>>3)&1), elem = (k&7) + 8*(k>>4)
//   B (32Kx16N): lane = n + 16*((k>>3)&1), elem = (k&7) + 8*(k>>4)
// Block = one 32-row chunk x one 256-column strip.
// ---------------------------------------------------------------------------
__global__ __launch_bounds__(256) void convert_kernel(const float* __restrict__ X,
                                                      const float* __restrict__ Tg,
                                                      __bf16* __restrict__ TbA,
                                                      __bf16* __restrict__ Phi,
                                                      __bf16* __restrict__ Plo,
                                                      float* __restrict__ ws) {
    const int chunk = blockIdx.x >> 3;            // 0..127
    const int strip = blockIdx.x & 7;             // 0..7
    const int tid   = threadIdx.x;
    const int col   = strip * 256 + tid;
    const int tile  = col >> 4;
    const int mn    = col & 15;
    const size_t cb = (size_t)chunk * (NTILE * TILE_E) + (size_t)tile * TILE_E;

    float s = 0.0f;
#pragma unroll 8
    for (int kk = 0; kk < KC; ++kk) {
        const int row = chunk * KC + kk;
        const float tv = Tg[(size_t)row * Ln + col];
        const float xv = X [(size_t)row * Ln + col];
        const int jA = (kk & 7) + ((kk >> 4) << 3);
        const int hA = (kk >> 3) & 1;
        const int jB = (kk & 7) + ((kk >> 4) << 3);
        const int hB = (kk >> 3) & 1;
        TbA[cb + (mn + 16 * hA) * 16 + jA] = (__bf16)tv;
        const float  p  = 1.0f / (1.0f + expf(-xv));
        const __bf16 hi = (__bf16)p;
        const size_t ob = cb + (mn + 16 * hB) * 16 + jB;
        Phi[ob] = hi;
        Plo[ob] = (__bf16)(p - (float)hi);
        s += p;
    }
    atomicAdd(&ws[16 + col], s);   // 128 adds per column total
}

// One block per row: focal partial sum + row stats (f64 scalar accumulators).
__global__ __launch_bounds__(256) void focal_row_kernel(const float* __restrict__ X,
                                                        const float* __restrict__ Tg,
                                                        double* __restrict__ wsd) {
    const int row = blockIdx.x;
    const int tid = threadIdx.x;
    const float* x = X + (size_t)row * Ln;
    const float* t = Tg + (size_t)row * Ln;

    float fsum = 0.0f, np = 0.0f, nt = 0.0f;
#pragma unroll
    for (int j = 0; j < Ln / 256; ++j) {
        const int c = tid + j * 256;
        const float xv = x[c];
        const float tv = t[c];
        const float p = 1.0f / (1.0f + expf(-xv));
        const float bce = fmaxf(xv, 0.0f) - xv * tv + log1pf(expf(-fabsf(xv)));
        const float om = 1.0f - p;
        fsum += ALPHA_C * om * om * bce;
        np += p * p;
        nt += tv;
    }

    __shared__ float r0[256], r1[256], r2[256];
    r0[tid] = fsum; r1[tid] = np; r2[tid] = nt;
    __syncthreads();
    for (int s = 128; s > 0; s >>= 1) {
        if (tid < s) { r0[tid] += r0[tid + s]; r1[tid] += r1[tid + s]; r2[tid] += r2[tid + s]; }
        __syncthreads();
    }
    if (tid == 0) {
        atomicAdd(&wsd[0], (double)r0[0]);
        atomicAdd(&wsd[1], (double)r1[0]);
        atomicAdd(&wsd[2], (double)(r2[0] * r1[0]));
    }
}

// ---------------------------------------------------------------------------
// ||T^T P||_F^2 : 128x128 macro tile, K-chunks of 32.
// Staging = pure async DMA (panels are contiguous 8 KB in fragment order),
// double-buffered; compute = 16 ds_load_b128 + 16 WMMA per wave per chunk.
// ---------------------------------------------------------------------------
__global__ __launch_bounds__(256) void frob_gemm_kernel(const __bf16* __restrict__ TbA,
                                                        const __bf16* __restrict__ Phi,
                                                        const __bf16* __restrict__ Plo,
                                                        float* __restrict__ ws) {
    __shared__ __align__(16) __bf16 sA [2][8 * TILE_E];   // 2 x 8 KB
    __shared__ __align__(16) __bf16 sBh[2][8 * TILE_E];
    __shared__ __align__(16) __bf16 sBl[2][8 * TILE_E];

    const int tid  = threadIdx.x;
    const int lane = tid & 31;
    const int wave = tid >> 5;
    const int bk   = blockIdx.x & 15;   // row-tile group (k of M)
    const int bl   = blockIdx.x >> 4;   // col-tile group (l of M)
    const int wr   = wave >> 2;         // 0..1
    const int wc   = wave & 3;          // 0..3

    // async copy geometry: each thread moves 32 contiguous bytes per panel
    const uint32_t aD = (uint32_t)(size_t)&sA [0][0] + tid * 32;
    const uint32_t hD = (uint32_t)(size_t)&sBh[0][0] + tid * 32;
    const uint32_t lD = (uint32_t)(size_t)&sBl[0][0] + tid * 32;
    const uint64_t chunkStride = (uint64_t)NTILE * TILE_E * 2;   // 128 KB... 128*1024 B
    const uint64_t aS = (uint64_t)(size_t)TbA + (uint64_t)(8 * bk) * (TILE_E * 2) + tid * 32;
    const uint64_t hS = (uint64_t)(size_t)Phi + (uint64_t)(8 * bl) * (TILE_E * 2) + tid * 32;
    const uint64_t lS = (uint64_t)(size_t)Plo + (uint64_t)(8 * bl) * (TILE_E * 2) + tid * 32;
    const uint32_t bufB = 8 * TILE_E * 2;   // 8192 bytes per buffer

    auto issue_chunk = [&](int buf, int chunk) {
        const uint64_t off = (uint64_t)chunk * chunkStride;
        const uint32_t ad = aD + buf * bufB;
        const uint32_t hd = hD + buf * bufB;
        const uint32_t ld = lD + buf * bufB;
        const uint64_t as = aS + off, hs = hS + off, ls = lS + off;
        asm volatile("global_load_async_to_lds_b128 %0, %1, off\n\t"
                     "global_load_async_to_lds_b128 %0, %1, off offset:16"
                     :: "v"(ad), "v"(as) : "memory");
        asm volatile("global_load_async_to_lds_b128 %0, %1, off\n\t"
                     "global_load_async_to_lds_b128 %0, %1, off offset:16"
                     :: "v"(hd), "v"(hs) : "memory");
        asm volatile("global_load_async_to_lds_b128 %0, %1, off\n\t"
                     "global_load_async_to_lds_b128 %0, %1, off offset:16"
                     :: "v"(ld), "v"(ls) : "memory");
    };

    v8f acc[4][2] = {};
    issue_chunk(0, 0);

    for (int c = 0; c < NCHUNK; ++c) {
        // drain own async copies, then barrier -> whole 24 KB stage visible
        asm volatile("s_wait_asynccnt 0x0" ::: "memory");
        __syncthreads();
        if (c + 1 < NCHUNK) issue_chunk((c + 1) & 1, c + 1);

        const int buf = c & 1;
        union V16 { uint4 q[2]; v16bf v; };
        V16 aF[4], bh[2], blo[2];
#pragma unroll
        for (int r = 0; r < 4; ++r) {
            const __bf16* p = &sA[buf][(4 * wr + r) * TILE_E + lane * 16];
            aF[r].q[0] = *(const uint4*)p;
            aF[r].q[1] = *(const uint4*)(p + 8);
        }
#pragma unroll
        for (int cc = 0; cc < 2; ++cc) {
            const __bf16* ph = &sBh[buf][(2 * wc + cc) * TILE_E + lane * 16];
            bh[cc].q[0] = *(const uint4*)ph;
            bh[cc].q[1] = *(const uint4*)(ph + 8);
            const __bf16* pl = &sBl[buf][(2 * wc + cc) * TILE_E + lane * 16];
            blo[cc].q[0] = *(const uint4*)pl;
            blo[cc].q[1] = *(const uint4*)(pl + 8);
        }
#pragma unroll
        for (int r = 0; r < 4; ++r)
#pragma unroll
            for (int cc = 0; cc < 2; ++cc) {
                acc[r][cc] = __builtin_amdgcn_wmma_f32_16x16x32_bf16(
                    false, aF[r].v, false, bh[cc].v, (short)0, acc[r][cc], false, false);
                acc[r][cc] = __builtin_amdgcn_wmma_f32_16x16x32_bf16(
                    false, aF[r].v, false, blo[cc].v, (short)0, acc[r][cc], false, false);
            }
    }

    // ---- Frobenius partial: sum of squares of accumulators ----
    float ssq = 0.0f;
#pragma unroll
    for (int r = 0; r < 4; ++r)
#pragma unroll
        for (int cc = 0; cc < 2; ++cc)
#pragma unroll
            for (int e = 0; e < 8; ++e)
                ssq += acc[r][cc][e] * acc[r][cc][e];

    __syncthreads();
    float* red = reinterpret_cast<float*>(&sA[0][0]);
    red[tid] = ssq;
    __syncthreads();
    for (int s = 128; s > 0; s >>= 1) {
        if (tid < s) red[tid] += red[tid + s];
        __syncthreads();
    }
    if (tid == 0) atomicAdd((double*)ws + 4, (double)red[0]);
}

__global__ __launch_bounds__(256) void finalize_kernel(const float* __restrict__ ws,
                                                       float* __restrict__ out) {
    const int tid = threadIdx.x;
    const double* wsd = (const double*)ws;
    const float* colsum = ws + 16;
    double s = 0.0;
#pragma unroll
    for (int j = 0; j < Ln / 256; ++j) {
        const double v = (double)colsum[tid + j * 256];
        s += v * v;
    }
    __shared__ double red[256];
    red[tid] = s;
    __syncthreads();
    for (int st = 128; st > 0; st >>= 1) {
        if (tid < st) red[tid] += red[tid + st];
        __syncthreads();
    }
    if (tid == 0) {
        const double ssum = red[0];                  // ||s||^2
        const double focal_mean = wsd[0] / (double)((long long)Bn * (long long)Ln);
        const double numer = (ssum - wsd[1]) - (wsd[4] - wsd[2]);
        const double contrastive = numer / ((double)Bn * (double)(Bn - 1));
        out[0] = (float)(focal_mean + contrastive);
    }
}

extern "C" void kernel_launch(void* const* d_in, const int* in_sizes, int n_in,
                              void* d_out, int out_size, void* d_ws, size_t ws_size,
                              hipStream_t stream) {
    (void)in_sizes; (void)n_in; (void)out_size; (void)ws_size;  // needs ~50.4 MB ws
    const float* X  = (const float*)d_in[0];   // inputs  [B, L] f32
    const float* Tg = (const float*)d_in[1];   // targets [B, L] f32 (0/1)
    float* out = (float*)d_out;                // scalar f32
    float* ws  = (float*)d_ws;
    __bf16* TbA = (__bf16*)((char*)d_ws + OFF_TB);
    __bf16* Phi = (__bf16*)((char*)d_ws + OFF_PH);
    __bf16* Plo = (__bf16*)((char*)d_ws + OFF_PL);

    zero_ws_kernel<<<(WS_FLOATS + 255) / 256, 256, 0, stream>>>(ws);
    convert_kernel<<<NCHUNK * 8, 256, 0, stream>>>(X, Tg, TbA, Phi, Plo, ws);
    focal_row_kernel<<<Bn, 256, 0, stream>>>(X, Tg, (double*)ws);
    frob_gemm_kernel<<<16 * 16, 256, 0, stream>>>(TbA, Phi, Plo, ws);
    finalize_kernel<<<1, 256, 0, stream>>>(ws, out);
}
